// LLaMAAttention_70128226009183
// MI455X (gfx1250) — compile-verified, matrix operand order checked by reading, not run on hardware
//
#include <hip/hip_runtime.h>
#include <stdint.h>
#include <stddef.h>

#define HIDDEN  4096
#define NHEADS  32
#define HEADD   128
#define BATCH   2
#define SEQ     2048
#define MROWS   (BATCH * SEQ)   /* 4096 */

typedef __attribute__((ext_vector_type(16))) __bf16 v16bf;
typedef __attribute__((ext_vector_type(8)))  float  v8f;
typedef __attribute__((ext_vector_type(4)))  unsigned int v4u;
typedef __attribute__((ext_vector_type(8)))  int    v8i;
typedef __attribute__((ext_vector_type(4)))  int    v4i;

#if defined(__has_builtin)
#  if __has_builtin(__builtin_amdgcn_tensor_load_to_lds)
#    define USE_TDM 1
#  endif
#endif

union FragBF {
    v16bf    v;
    uint32_t u[8];
};

__device__ __forceinline__ uint16_t f2bf(float f) {
    uint32_t u = __float_as_uint(f);
    u += 0x7FFFu + ((u >> 16) & 1u);      // round-to-nearest-even
    return (uint16_t)(u >> 16);
}
__device__ __forceinline__ float bf2f(uint16_t h) {
    return __uint_as_float(((uint32_t)h) << 16);
}

// K-dim pair base for 16-bit A/B WMMA fragments (ISA 7.12.2):
// lanes 0-15 hold K = 0..7 / 16..23, lanes 16-31 hold K = 8..15 / 24..31.
__device__ __forceinline__ int kpair(int v, int half) {
    int b = (v < 4) ? (2 * v) : (16 + 2 * (v - 4));
    return b + 8 * half;
}

// Byte address of a __shared__ object inside the wave's LDS space
// (generic -> LDS addrspace cast, then truncate to the 32-bit LDS offset).
__device__ __forceinline__ uint32_t lds_off_u32(const void* p) {
    return (uint32_t)(uintptr_t)(__attribute__((address_space(3))) const void*)p;
}

// ---------------------------------------------------------------------------
// Kernel 1: fp32 -> bf16 cast (vectorized x4)
// ---------------------------------------------------------------------------
__global__ void cast_f32_to_bf16(const float* __restrict__ in,
                                 uint16_t* __restrict__ out, int n4) {
    int i = blockIdx.x * blockDim.x + threadIdx.x;
    if (i >= n4) return;
    float4 f = reinterpret_cast<const float4*>(in)[i];
    uint32_t lo = (uint32_t)f2bf(f.x) | ((uint32_t)f2bf(f.y) << 16);
    uint32_t hi = (uint32_t)f2bf(f.z) | ((uint32_t)f2bf(f.w) << 16);
    reinterpret_cast<uint2*>(out)[i] = make_uint2(lo, hi);
}

// ---------------------------------------------------------------------------
// Kernel 2: bf16 WMMA GEMM  C[M x N] = A[M x K] * W[N x K]^T   (M=N=K=4096)
//   mode 0: store fp32 row-major into outF (final projection -> d_out)
//   mode 1: store bf16 into outQ with [B, H, S, D] layout (Q/K/V projection)
// Each wave: 16 rows x 64 cols, 8 waves per block, explicit double buffering.
// ---------------------------------------------------------------------------
__global__ __launch_bounds__(256)
void gemm_bf16_wmma(const uint16_t* __restrict__ A,
                    const uint16_t* __restrict__ W,
                    float* __restrict__ outF,
                    uint16_t* __restrict__ outQ,
                    int mode) {
    const int Kdim = 4096, Ndim = 4096;
    const int lane = threadIdx.x & 31;
    const int wv   = threadIdx.x >> 5;
    const int half = lane >> 4;
    const int l15  = lane & 15;
    const int m0   = (blockIdx.y * 8 + wv) * 16;
    const int n0   = blockIdx.x * 64;

    v8f acc[4];
#pragma unroll
    for (int t = 0; t < 4; ++t)
#pragma unroll
        for (int r = 0; r < 8; ++r) acc[t][r] = 0.0f;

    const uint16_t* arowBase = A + (size_t)(m0 + l15) * Kdim;

    auto loadA = [&](FragBF& a, int kb) {
        const uint16_t* arow = arowBase + kb;
#pragma unroll
        for (int v = 0; v < 8; ++v)
            a.u[v] = *reinterpret_cast<const uint32_t*>(arow + kpair(v, half));
        __builtin_prefetch(arow + 96, 0, 1);   // global_prefetch_b8
    };
    auto loadB = [&](FragBF (&b)[4], int kb) {
#pragma unroll
        for (int t = 0; t < 4; ++t) {
            const uint16_t* wrow = W + (size_t)(n0 + t * 16 + l15) * Kdim + kb;
#pragma unroll
            for (int v = 0; v < 8; ++v)
                b[t].u[v] =
                    *reinterpret_cast<const uint32_t*>(wrow + kpair(v, half));
        }
    };

    // Ping/pong fragment sets, manually unrolled x2 so no register copies.
    FragBF aA, aB, bA[4], bB[4];
    loadA(aA, 0);
    loadB(bA, 0);
    for (int kb = 0; kb < Kdim; kb += 64) {
        loadA(aB, kb + 32);
        loadB(bB, kb + 32);
#pragma unroll
        for (int t = 0; t < 4; ++t)
            acc[t] = __builtin_amdgcn_wmma_f32_16x16x32_bf16(
                false, aA.v, false, bA[t].v, (short)0, acc[t], false, false);
        int k2 = (kb + 64 < Kdim) ? (kb + 64) : kb;   // clamped tail prefetch
        loadA(aA, k2);
        loadB(bA, k2);
#pragma unroll
        for (int t = 0; t < 4; ++t)
            acc[t] = __builtin_amdgcn_wmma_f32_16x16x32_bf16(
                false, aB.v, false, bB[t].v, (short)0, acc[t], false, false);
    }

    if (mode == 0) {
#pragma unroll
        for (int t = 0; t < 4; ++t) {
            int n = n0 + t * 16 + l15;
#pragma unroll
            for (int r = 0; r < 8; ++r) {
                int m = m0 + r + 8 * half;
                outF[(size_t)m * Ndim + n] = acc[t][r];
            }
        }
    } else {
#pragma unroll
        for (int t = 0; t < 4; ++t) {
            int e    = n0 + t * 16 + l15;   // hidden-dim index
            int head = e >> 7;              // /128
            int dh   = e & 127;
#pragma unroll
            for (int r = 0; r < 8; ++r) {
                int m  = m0 + r + 8 * half; // global row = b*SEQ + s
                int bb = m >> 11;           // /2048
                int ss = m & 2047;
                outQ[(((size_t)bb * NHEADS + head) * SEQ + ss) * HEADD + dh] =
                    f2bf(acc[t][r]);
            }
        }
    }
}

// ---------------------------------------------------------------------------
// Kernel 3: RoPE on Q and K, layout [B, H, S, D] bf16, cos/sin [S, 64] fp32
// ---------------------------------------------------------------------------
__global__ void rope_kernel(uint16_t* __restrict__ Q, uint16_t* __restrict__ K,
                            const float* __restrict__ cosT,
                            const float* __restrict__ sinT) {
    size_t i = (size_t)blockIdx.x * blockDim.x + threadIdx.x;
    const size_t total = (size_t)BATCH * NHEADS * SEQ * (HEADD / 2);
    if (i >= total) return;
    int    d2   = (int)(i & 63);
    size_t sh   = i >> 6;              // (b*NHEADS + h)*SEQ + s
    int    s    = (int)(sh & (SEQ - 1));
    size_t base = sh * HEADD;
    float c  = cosT[(size_t)s * 64 + d2];
    float sn = sinT[(size_t)s * 64 + d2];

    float q1 = bf2f(Q[base + d2]),      q2 = bf2f(Q[base + 64 + d2]);
    Q[base + d2]      = f2bf(q1 * c - q2 * sn);
    Q[base + 64 + d2] = f2bf(q2 * c + q1 * sn);

    float k1 = bf2f(K[base + d2]),      k2 = bf2f(K[base + 64 + d2]);
    K[base + d2]      = f2bf(k1 * c - k2 * sn);
    K[base + 64 + d2] = f2bf(k2 * c + k1 * sn);
}

// ---------------------------------------------------------------------------
// Kernel 4: causal flash attention.
//   grid = (SEQ/64, BATCH*NHEADS), block = 128 (4 waves).
//   K tile staged by the Tensor Data Mover (TDM) from wave 0 (TENSORcnt);
//   V tile staged by async global->LDS b128 copies (ASYNCcnt). Both tiles are
//   row-major [key][d] in LDS. Online softmax fp32, P/V bf16 WMMA.
// ---------------------------------------------------------------------------
__global__ __launch_bounds__(128)
void attn_kernel(const uint16_t* __restrict__ Qb,
                 const uint16_t* __restrict__ Kb,
                 const uint16_t* __restrict__ Vb,
                 uint16_t* __restrict__ Ob) {
    __shared__ uint16_t sK[32 * 128];       // [key][d]
    __shared__ uint16_t sV[32 * 128];       // [key][d]
    __shared__ uint16_t sP[4][16 * 32];     // per-wave P repack buffer

    const int tid  = threadIdx.x;
    const int lane = tid & 31;
    const int wv   = tid >> 5;
    const int half = lane >> 4;
    const int l15  = lane & 15;
    const int bh   = blockIdx.y;            // b*NHEADS + h
    const int b    = bh >> 5;
    const int h    = bh & 31;
    const int q0   = blockIdx.x * 64 + wv * 16;
    const size_t headBase = (size_t)bh * SEQ * HEADD;
    const float  scale    = 0.08838834764831845f;  // 1/sqrt(128)

    // Preload Q fragments (16 rows x 128 d, four 32-wide k-chunks).
    FragBF qf[4];
    {
        const uint16_t* qrow = Qb + headBase + (size_t)(q0 + l15) * HEADD;
#pragma unroll
        for (int dc = 0; dc < 4; ++dc)
#pragma unroll
            for (int v = 0; v < 8; ++v)
                qf[dc].u[v] = *reinterpret_cast<const uint32_t*>(
                    qrow + dc * 32 + kpair(v, half));
    }

    v8f acc[8];
#pragma unroll
    for (int t = 0; t < 8; ++t)
#pragma unroll
        for (int r = 0; r < 8; ++r) acc[t][r] = 0.0f;
    float mrow[8], lrow[8];
#pragma unroll
    for (int r = 0; r < 8; ++r) { mrow[r] = -1.0e30f; lrow[r] = 0.0f; }

    const uint32_t sK_off = lds_off_u32(sK);
    const uint32_t sV_off = lds_off_u32(sV);

    const int nkb = 2 * blockIdx.x + 2;     // causal: key blocks up to q0_max
    for (int ib = 0; ib < nkb; ++ib) {
        const int kb = ib * 32;

        // ---- stage K tile via the Tensor Data Mover (wave 0 only) ----
        const uint16_t* kgp = Kb + headBase + (size_t)kb * HEADD;
#if defined(USE_TDM)
        if (wv == 0) {
            uint64_t ga = (uint64_t)(uintptr_t)kgp;
            v4u g0;
            g0[0] = 1u;                                  // count=1, user mode
            g0[1] = sK_off;                              // lds_addr
            g0[2] = (uint32_t)(ga & 0xFFFFFFFFu);        // global_addr[31:0]
            g0[3] = (uint32_t)((ga >> 32) & 0x01FFFFFFu) // global_addr[56:32]
                    | (2u << 30);                        // type=2 ("image")
            v8i g1;
            g1[0] = (int)(1u << 16);        // data_size=1 (2 bytes/elem)
            g1[1] = (int)(128u << 16);      // tensor_dim0[15:0]=128 @bits63:48
            g1[2] = (int)(2048u << 16);     // tensor_dim1[15:0]=2048 @bits95:80
            g1[3] = (int)(128u << 16);      // tile_dim0=128 @bits127:112
            g1[4] = 32;                     // tile_dim1=32  @bits143:128
            g1[5] = 128;                    // tensor_dim0_stride=128
            g1[6] = 0;
            g1[7] = 0;
            v4i z4 = {0, 0, 0, 0};          // <=2D tensor: groups 2/3 unused
            v8i z8 = {0, 0, 0, 0, 0, 0, 0, 0};
            __builtin_amdgcn_tensor_load_to_lds(g0, g1, z4, z4, z8, 0);
            __builtin_amdgcn_s_wait_tensorcnt(0);
        }
#else
        {
            const uint32_t* Kg = reinterpret_cast<const uint32_t*>(kgp);
            for (int i = tid; i < 32 * 64; i += 128)
                reinterpret_cast<uint32_t*>(sK)[i] = Kg[i];
        }
#endif
        // ---- stage V tile via async global->LDS b128 copies ----
        {
            const char* vg =
                (const char*)(Vb + headBase + (size_t)kb * HEADD);
#pragma unroll
            for (int it = 0; it < 4; ++it) {
                int i = tid + it * 128;                 // 512 b128 chunks
                uint64_t ga = (uint64_t)(uintptr_t)(vg + i * 16);
                uint32_t la = sV_off + (uint32_t)(i * 16);
                asm volatile("global_load_async_to_lds_b128 %0, %1, off"
                             :: "v"(la), "v"(ga) : "memory");
            }
            asm volatile("s_wait_asynccnt 0x0" ::: "memory");
        }
        __syncthreads();

        if (kb <= q0 + 15) {   // wave-uniform causal skip
            // ---- scores S = Q K^T (two 16-key tiles) ----
            v8f st[2];
#pragma unroll
            for (int t = 0; t < 2; ++t)
#pragma unroll
                for (int r = 0; r < 8; ++r) st[t][r] = 0.0f;
#pragma unroll
            for (int t = 0; t < 2; ++t) {
#pragma unroll
                for (int dc = 0; dc < 4; ++dc) {
                    FragBF kf;
                    const uint16_t* krow = &sK[(t * 16 + l15) * 128 + dc * 32];
#pragma unroll
                    for (int v = 0; v < 8; ++v)
                        kf.u[v] = *reinterpret_cast<const uint32_t*>(
                            krow + kpair(v, half));
                    st[t] = __builtin_amdgcn_wmma_f32_16x16x32_bf16(
                        false, qf[dc].v, false, kf.v, (short)0, st[t],
                        false, false);
                }
            }
            // ---- scale + causal mask ----
            float pm[2][8];
#pragma unroll
            for (int t = 0; t < 2; ++t)
#pragma unroll
                for (int r = 0; r < 8; ++r) {
                    int q   = q0 + r + 8 * half;
                    int key = kb + t * 16 + l15;
                    float s = st[t][r] * scale;
                    pm[t][r] = (key <= q) ? s : -1.0e30f;
                }
            // ---- online softmax (row reductions inside 16-lane halves) ----
#pragma unroll
            for (int r = 0; r < 8; ++r) {
                float mx = fmaxf(pm[0][r], pm[1][r]);
#pragma unroll
                for (int off = 8; off > 0; off >>= 1)
                    mx = fmaxf(mx, __shfl_xor(mx, off, 32));
                float mnew  = fmaxf(mrow[r], mx);
                float alpha = __expf(mrow[r] - mnew);
                float p0 = __expf(pm[0][r] - mnew);
                float p1 = __expf(pm[1][r] - mnew);
                float rs = p0 + p1;
#pragma unroll
                for (int off = 8; off > 0; off >>= 1)
                    rs += __shfl_xor(rs, off, 32);
                lrow[r] = lrow[r] * alpha + rs;
                mrow[r] = mnew;
#pragma unroll
                for (int t = 0; t < 8; ++t) acc[t][r] *= alpha;
                pm[0][r] = p0;
                pm[1][r] = p1;
            }
            // ---- repack P (D-layout) -> A-layout via per-wave LDS tile ----
#pragma unroll
            for (int t = 0; t < 2; ++t)
#pragma unroll
                for (int r = 0; r < 8; ++r)
                    sP[wv][(r + 8 * half) * 32 + t * 16 + l15] = f2bf(pm[t][r]);
            FragBF pa;
#pragma unroll
            for (int v = 0; v < 8; ++v)
                pa.u[v] = *reinterpret_cast<const uint32_t*>(
                    &sP[wv][l15 * 32 + kpair(v, half)]);
            // ---- O += P V : B-frag pairs stride over keys (row-major sV) ---
#pragma unroll
            for (int dt = 0; dt < 8; ++dt) {
                int d = dt * 16 + l15;
                FragBF vf;
#pragma unroll
                for (int v = 0; v < 8; ++v) {
                    int kk = kpair(v, half);
                    uint32_t lo = sV[(kk + 0) * 128 + d];
                    uint32_t hi = sV[(kk + 1) * 128 + d];
                    vf.u[v] = lo | (hi << 16);
                }
                acc[dt] = __builtin_amdgcn_wmma_f32_16x16x32_bf16(
                    false, pa.v, false, vf.v, (short)0, acc[dt], false, false);
            }
        }
        __syncthreads();
    }

    // Epilogue: normalize and store O as bf16 [B, S, HIDDEN].
#pragma unroll
    for (int r = 0; r < 8; ++r) {
        float inv = 1.0f / lrow[r];
        int q = q0 + r + 8 * half;
#pragma unroll
        for (int dt = 0; dt < 8; ++dt) {
            int d = dt * 16 + l15;
            Ob[((size_t)b * SEQ + q) * HIDDEN + h * HEADD + d] =
                f2bf(acc[dt][r] * inv);
        }
    }
}

// ---------------------------------------------------------------------------
// Launch: cast -> QKV GEMMs -> RoPE -> attention -> output projection
// ---------------------------------------------------------------------------
extern "C" void kernel_launch(void* const* d_in, const int* in_sizes, int n_in,
                              void* d_out, int out_size, void* d_ws,
                              size_t ws_size, hipStream_t stream) {
    (void)in_sizes; (void)n_in; (void)out_size; (void)ws_size;

    const float* x    = (const float*)d_in[0];
    const float* cosT = (const float*)d_in[1];
    const float* sinT = (const float*)d_in[2];
    const float* wq   = (const float*)d_in[3];
    const float* wk   = (const float*)d_in[4];
    const float* wv   = (const float*)d_in[5];
    const float* wo   = (const float*)d_in[6];
    float* out = (float*)d_out;

    const size_t U = (size_t)HIDDEN * MROWS * sizeof(uint16_t); // 32 MiB unit
    char* ws = (char*)d_ws;
    uint16_t* xb  = (uint16_t*)(ws + 0 * U);   // bf16 x; reused later as O
    uint16_t* wqb = (uint16_t*)(ws + 1 * U);
    uint16_t* wkb = (uint16_t*)(ws + 2 * U);
    uint16_t* wvb = (uint16_t*)(ws + 3 * U);
    uint16_t* wob = (uint16_t*)(ws + 4 * U);
    uint16_t* Qb  = (uint16_t*)(ws + 5 * U);
    uint16_t* Kb  = (uint16_t*)(ws + 6 * U);
    uint16_t* Vb  = (uint16_t*)(ws + 7 * U);
    uint16_t* Ob  = xb;   // alias: x is dead after the QKV projections

    // 1) casts (each array: 4096*4096 fp32 -> bf16)
    const int n4 = (MROWS * HIDDEN) / 4;       // 4,194,304
    const int cgrid = n4 / 256;
    cast_f32_to_bf16<<<cgrid, 256, 0, stream>>>(x,  xb,  n4);
    cast_f32_to_bf16<<<cgrid, 256, 0, stream>>>(wq, wqb, n4);
    cast_f32_to_bf16<<<cgrid, 256, 0, stream>>>(wk, wkb, n4);
    cast_f32_to_bf16<<<cgrid, 256, 0, stream>>>(wv, wvb, n4);
    cast_f32_to_bf16<<<cgrid, 256, 0, stream>>>(wo, wob, n4);

    // 2) QKV projections (store directly in [B,H,S,D] bf16)
    dim3 ggrid(HIDDEN / 64, MROWS / 128);      // (64, 32)
    gemm_bf16_wmma<<<ggrid, 256, 0, stream>>>(xb, wqb, nullptr, Qb, 1);
    gemm_bf16_wmma<<<ggrid, 256, 0, stream>>>(xb, wkb, nullptr, Kb, 1);
    gemm_bf16_wmma<<<ggrid, 256, 0, stream>>>(xb, wvb, nullptr, Vb, 1);

    // 3) RoPE on Q and K
    {
        const size_t total = (size_t)BATCH * NHEADS * SEQ * (HEADD / 2);
        int rgrid = (int)((total + 255) / 256);
        rope_kernel<<<rgrid, 256, 0, stream>>>(Qb, Kb, cosT, sinT);
    }

    // 4) causal flash attention -> Ob (bf16, [B,S,HIDDEN])
    {
        dim3 agrid(SEQ / 64, BATCH * NHEADS);  // (32, 64)
        attn_kernel<<<agrid, 128, 0, stream>>>(Qb, Kb, Vb, Ob);
    }

    // 5) output projection -> fp32 d_out
    gemm_bf16_wmma<<<ggrid, 256, 0, stream>>>(Ob, wob, out, nullptr, 0);
}
